// FlaxRwkvSelfAttention_70514773065874
// MI455X (gfx1250) — matrix operand order repaired and checked by reading, not compiled, not run on hardware
//
#include <hip/hip_runtime.h>

#define HDIM 1024
#define TDIM 4096
#define BDIM 4
#define MDIM (BDIM * TDIM)   // 16384 rows for all GEMMs
#define BK   32
#define LDT  40              // LDS row stride in bf16 elems (80B) -> bank-conflict-free b128 frags

typedef __attribute__((ext_vector_type(16))) __bf16 bf16x16;
typedef __attribute__((ext_vector_type(8)))  __bf16 bf16x8;
typedef __attribute__((ext_vector_type(8)))  float  f32x8;

// ---------------- weight convert + transpose (fp32 KxN -> bf16 NxK) ----------------
__global__ __launch_bounds__(256) void wt_kernel(const float* __restrict__ W0,
                                                 const float* __restrict__ W1,
                                                 const float* __restrict__ W2,
                                                 const float* __restrict__ W3,
                                                 __bf16* __restrict__ dst) {
    __shared__ float tile[32][33];
    const float* src = (blockIdx.z == 0) ? W0 : (blockIdx.z == 1) ? W1
                     : (blockIdx.z == 2) ? W2 : W3;
    __bf16* out = dst + (size_t)blockIdx.z * HDIM * HDIM;
    const int bx = blockIdx.x * 32;  // n tile
    const int by = blockIdx.y * 32;  // k tile
    const int tx = threadIdx.x, ty = threadIdx.y;  // 32 x 8
#pragma unroll
    for (int r = 0; r < 4; ++r)
        tile[ty + r * 8][tx] = src[(size_t)(by + ty + r * 8) * HDIM + bx + tx];
    __syncthreads();
#pragma unroll
    for (int r = 0; r < 4; ++r)
        out[(size_t)(bx + ty + r * 8) * HDIM + by + tx] = (__bf16)tile[tx][ty + r * 8];
}

// ---------------- fused time-shift mix -> bf16 k_in/v_in/r_in ----------------
__global__ __launch_bounds__(256) void mix_kernel(const float* __restrict__ hidden,
                                                  const float* __restrict__ tmk,
                                                  const float* __restrict__ tmv,
                                                  const float* __restrict__ tmr,
                                                  __bf16* __restrict__ kin,
                                                  __bf16* __restrict__ vin,
                                                  __bf16* __restrict__ rin) {
    size_t i = (size_t)blockIdx.x * 256 + threadIdx.x;
    int h = (int)(i & (HDIM - 1));
    int t = (int)((i >> 10) & (TDIM - 1));
    float x = hidden[i];
    float s = (t == 0) ? 0.0f : hidden[i - HDIM];
    float mk = tmk[h], mv = tmv[h], mr = tmr[h];
    kin[i] = (__bf16)(x * mk + s * (1.0f - mk));
    vin[i] = (__bf16)(x * mv + s * (1.0f - mv));
    rin[i] = (__bf16)(x * mr + s * (1.0f - mr));
}

// ---------------- WMMA fragment loads (per ISA 16-bit layout tables) ----------------
__device__ __forceinline__ bf16x16 frag_a(const __bf16* lds, int row0, int lane) {
    int r = lane & 15, hf = lane >> 4;
    const __bf16* p = lds + (row0 + r) * LDT;
    bf16x8 lo = *(const bf16x8*)(p + hf * 8);        // K = hf*8 .. hf*8+7
    bf16x8 hi = *(const bf16x8*)(p + 16 + hf * 8);   // K = 16+hf*8 .. +7
    return __builtin_shufflevector(lo, hi, 0,1,2,3,4,5,6,7,8,9,10,11,12,13,14,15);
}
__device__ __forceinline__ bf16x16 frag_b(const __bf16* lds, int col0, int lane) {
    int c = lane & 15, hf = lane >> 4;
    const __bf16* p = lds + (col0 + c) * LDT;  // K = hf*8 .. +7, then 16+hf*8 .. +7
    bf16x8 lo = *(const bf16x8*)(p + hf * 8);
    bf16x8 hi = *(const bf16x8*)(p + 16 + hf * 8);
    return __builtin_shufflevector(lo, hi, 0,1,2,3,4,5,6,7,8,9,10,11,12,13,14,15);
}

// ---------------- bf16 WMMA GEMM: C(MDIMxH) = A(MDIMxH) * Wt(HxH, N-major) ----------
// mode 0: store fp32 to outF; mode 1: store sigmoid() as bf16 to outB
__global__ __launch_bounds__(256) void gemm_kernel(const __bf16* __restrict__ A,
                                                   const __bf16* __restrict__ Wt,
                                                   float* __restrict__ outF,
                                                   __bf16* __restrict__ outB,
                                                   int mode) {
    __shared__ __bf16 lsA[128 * LDT];
    __shared__ __bf16 lsB[128 * LDT];
    const int tid  = threadIdx.x;
    const int lane = tid & 31;
    const int wave = tid >> 5;
    const int wm = (wave >> 2) * 64;   // wave row offset (0/64)
    const int wn = (wave & 3) * 32;    // wave col offset (0..96)
    const int blkM = blockIdx.y * 128;
    const int blkN = blockIdx.x * 128;

    // tile staging: 512 16B-chunks per tile, 2 per thread
    const int c0 = tid, c1 = tid + 256;
    const int ar0 = c0 >> 2, ac0 = c0 & 3;
    const int ar1 = c1 >> 2, ac1 = c1 & 3;

    f32x8 acc[4][2] = {};
    const int KT = HDIM / BK;  // 32
    for (int kk = 0; kk < KT; ++kk) {
        const __bf16* gA = A  + (size_t)blkM * HDIM + kk * BK;
        const __bf16* gB = Wt + (size_t)blkN * HDIM + kk * BK;
        bf16x8 va0 = *(const bf16x8*)(gA + (size_t)ar0 * HDIM + ac0 * 8);
        bf16x8 va1 = *(const bf16x8*)(gA + (size_t)ar1 * HDIM + ac1 * 8);
        bf16x8 vb0 = *(const bf16x8*)(gB + (size_t)ar0 * HDIM + ac0 * 8);
        bf16x8 vb1 = *(const bf16x8*)(gB + (size_t)ar1 * HDIM + ac1 * 8);
        if (kk + 1 < KT) {  // uniform branch; pull next K-tile toward L2/L0
            __builtin_prefetch(gA + (size_t)ar0 * HDIM + BK, 0, 1);
            __builtin_prefetch(gB + (size_t)ar0 * HDIM + BK, 0, 1);
        }
        __syncthreads();
        *(bf16x8*)&lsA[ar0 * LDT + ac0 * 8] = va0;
        *(bf16x8*)&lsA[ar1 * LDT + ac1 * 8] = va1;
        *(bf16x8*)&lsB[ar0 * LDT + ac0 * 8] = vb0;
        *(bf16x8*)&lsB[ar1 * LDT + ac1 * 8] = vb1;
        __syncthreads();

        bf16x16 af[4];
        bf16x16 bfr[2];
#pragma unroll
        for (int i = 0; i < 4; ++i) af[i] = frag_a(lsA, wm + i * 16, lane);
#pragma unroll
        for (int j = 0; j < 2; ++j) bfr[j] = frag_b(lsB, wn + j * 16, lane);
#pragma unroll
        for (int i = 0; i < 4; ++i)
#pragma unroll
            for (int j = 0; j < 2; ++j)
                acc[i][j] = __builtin_amdgcn_wmma_f32_16x16x32_bf16(
                    false, af[i], false, bfr[j], (short)0, acc[i][j], false, false);
    }

    // epilogue: lane l, vgpr e -> M = e + 8*(l>=16), N = l&15
    const int hf = lane >> 4, cl = lane & 15;
#pragma unroll
    for (int i = 0; i < 4; ++i)
#pragma unroll
        for (int j = 0; j < 2; ++j)
#pragma unroll
            for (int e = 0; e < 8; ++e) {
                int row = blkM + wm + i * 16 + hf * 8 + e;
                int col = blkN + wn + j * 16 + cl;
                float v = acc[i][j][e];
                size_t idx = (size_t)row * HDIM + col;
                if (mode == 0) outF[idx] = v;
                else           outB[idx] = (__bf16)(1.0f / (1.0f + __expf(-v)));
            }
}

// ---------------- WKV scan: one lane per (b,h) channel, 8-deep load batching -------
__global__ __launch_bounds__(256) void wkv_kernel(const float* __restrict__ K,
                                                  const float* __restrict__ V,
                                                  const __bf16* __restrict__ R,
                                                  const float* __restrict__ time_decay,
                                                  const float* __restrict__ time_first,
                                                  __bf16* __restrict__ Aout) {
    int ch = blockIdx.x * 256 + threadIdx.x;   // 0..4095
    int b  = ch >> 10;
    int h  = ch & (HDIM - 1);
    const float td = -__expf(time_decay[h]);
    const float tf = time_first[h];
    size_t base = (size_t)b * TDIM * HDIM + h;
    float num = 0.0f, den = 0.0f, mx = -1e38f;
    for (int t0 = 0; t0 < TDIM; t0 += 8) {
        float kc[8], vc[8], rc[8];
#pragma unroll
        for (int u = 0; u < 8; ++u) {
            size_t idx = base + (size_t)(t0 + u) * HDIM;
            kc[u] = K[idx];
            vc[u] = V[idx];
            rc[u] = (float)R[idx];
        }
#pragma unroll
        for (int u = 0; u < 8; ++u) {
            float ck = kc[u], cv = vc[u];
            float mo  = fmaxf(mx, ck + tf);
            float e1  = __expf(mx - mo);
            float e2  = __expf(ck + tf - mo);
            float out = (e1 * num + e2 * cv) / (e1 * den + e2);
            float ms  = fmaxf(mx + td, ck);
            float e1s = __expf(mx + td - ms);
            float e2s = __expf(ck - ms);
            num = e1s * num + e2s * cv;
            den = e1s * den + e2s;
            mx  = ms;
            Aout[base + (size_t)(t0 + u) * HDIM] = (__bf16)(rc[u] * out);
        }
    }
}

// ---------------- host orchestration ----------------
extern "C" void kernel_launch(void* const* d_in, const int* in_sizes, int n_in,
                              void* d_out, int out_size, void* d_ws, size_t ws_size,
                              hipStream_t stream) {
    (void)in_sizes; (void)n_in; (void)out_size; (void)ws_size;
    const float* hidden = (const float*)d_in[0];
    const float* Wk = (const float*)d_in[1];
    const float* Wv = (const float*)d_in[2];
    const float* Wr = (const float*)d_in[3];
    const float* Wo = (const float*)d_in[4];
    const float* td  = (const float*)d_in[5];
    const float* tf  = (const float*)d_in[6];
    const float* tmk = (const float*)d_in[7];
    const float* tmv = (const float*)d_in[8];
    const float* tmr = (const float*)d_in[9];

    char* ws = (char*)d_ws;
    const size_t MB = 1024ull * 1024ull;
    __bf16* WT  = (__bf16*)(ws + 0);          //   8 MiB: 4x bf16 HxH (transposed)
    __bf16* kin = (__bf16*)(ws + 8  * MB);    //  32 MiB
    __bf16* vin = (__bf16*)(ws + 40 * MB);    //  32 MiB
    __bf16* rin = (__bf16*)(ws + 72 * MB);    //  32 MiB
    float*  Kb  = (float*) (ws + 104 * MB);   //  64 MiB
    float*  Vb  = (float*) (ws + 168 * MB);   //  64 MiB
    __bf16* Rb  = (__bf16*)(ws + 232 * MB);   //  32 MiB (sigmoid(r))
    __bf16* Ab  = kin;                        // reuse: r*wkv, kin consumed by then
    __bf16* WkT = WT;
    __bf16* WvT = WT + (size_t)HDIM * HDIM;
    __bf16* WrT = WT + (size_t)2 * HDIM * HDIM;
    __bf16* WoT = WT + (size_t)3 * HDIM * HDIM;

    {   // weights -> bf16 transposed
        dim3 tb(32, 8), tg(HDIM / 32, HDIM / 32, 4);
        wt_kernel<<<tg, tb, 0, stream>>>(Wk, Wv, Wr, Wo, WT);
    }
    mix_kernel<<<(MDIM * HDIM) / 256, 256, 0, stream>>>(hidden, tmk, tmv, tmr,
                                                        kin, vin, rin);
    dim3 gg(HDIM / 128, MDIM / 128);
    gemm_kernel<<<gg, 256, 0, stream>>>(kin, WkT, Kb, nullptr, 0);
    gemm_kernel<<<gg, 256, 0, stream>>>(vin, WvT, Vb, nullptr, 0);
    gemm_kernel<<<gg, 256, 0, stream>>>(rin, WrT, nullptr, Rb, 1);

    wkv_kernel<<<(BDIM * HDIM) / 256, 256, 0, stream>>>(Kb, Vb, Rb, td, tf, Ab);

    gemm_kernel<<<gg, 256, 0, stream>>>(Ab, WoT, (float*)d_out, nullptr, 0);
}
